// RoPEMultiheadAttention_72103910966038
// MI455X (gfx1250) — matrix operand order repaired and checked by reading, not compiled, not run on hardware
//
#include <hip/hip_runtime.h>
#include <hip/hip_bf16.h>
#include <math.h>

typedef _Float16 h16;
typedef __attribute__((ext_vector_type(16))) _Float16 v16h;
typedef __attribute__((ext_vector_type(8)))  _Float16 v8h;
typedef __attribute__((ext_vector_type(4)))  _Float16 v4h;
typedef __attribute__((ext_vector_type(8)))  float    v8f;

#define D_MODEL 1024
#define NHEADS  16
#define HDIM    64
#define BATCH   256
#define SEQ     128
#define MTOK    (BATCH*SEQ)   // 32768

// ---- CDNA5 async global->LDS copy (ASYNCcnt path), ISA §15.18.3 op 98 ----
typedef __attribute__((address_space(3))) void lds_void_t;

__device__ __forceinline__ void async_load_b128(const void* gaddr, void* lgen) {
  unsigned loff = (unsigned)(unsigned long long)(lds_void_t*)lgen; // LDS byte offset
  asm volatile("global_load_async_to_lds_b128 %0, %1, off"
               :: "v"(loff), "v"(gaddr) : "memory");
}
__device__ __forceinline__ void wait_async_zero() {
  asm volatile("s_wait_asynccnt 0" ::: "memory");
}

// ---------------------------------------------------------------- convert
__global__ void cvt_f32_to_f16(const float* __restrict__ src,
                               h16* __restrict__ dst, int n4) {
  int i = blockIdx.x * blockDim.x + threadIdx.x;
  if (i < n4) {
    float4 f = ((const float4*)src)[i];
    v4h o; o.x = (h16)f.x; o.y = (h16)f.y; o.z = (h16)f.z; o.w = (h16)f.w;
    ((v4h*)dst)[i] = o;
  }
}

// ---------------------------------------------------------------- GEMM
// C[m,n] = sum_k A[m,k] * W[n,k] + bias[n]
// MODE 1: write f16, scattered to [b*NHEADS+h][t][d]  (QKV)
// MODE 2: write f32 row-major [M,N]                   (output projection)
template <int MODE>
__global__ __launch_bounds__(256)
void gemm_wmma_f16(const h16* __restrict__ A,     // [M,K] row-major
                   const h16* __restrict__ W,     // [N,K] row-major
                   const float* __restrict__ bias,
                   void* __restrict__ outp,
                   int M, int N, int K) {
  __shared__ __align__(32) h16 As[2][128 * 32];   // double-buffered, 16KB
  __shared__ __align__(32) h16 Ws[2][128 * 32];   // double-buffered, 16KB

  const int tid  = threadIdx.x;
  const int wave = tid >> 5;
  const int lane = tid & 31;
  const int lh   = lane & 15;
  const int hi   = (lane >> 4) & 1;

  const int n0 = blockIdx.x * 128;
  const int m0 = blockIdx.y * 128;

  // per-thread async-load slots: rows r0 and r0+64, one 16B chunk each
  const int r0 = tid >> 2;
  const int c0 = (tid & 3) * 8;

  auto issue_tile = [&](int buf, int k0) {
    async_load_b128(&A[(size_t)(m0 + r0)      * K + k0 + c0], &As[buf][r0 * 32 + c0]);
    async_load_b128(&A[(size_t)(m0 + r0 + 64) * K + k0 + c0], &As[buf][(r0 + 64) * 32 + c0]);
    async_load_b128(&W[(size_t)(n0 + r0)      * K + k0 + c0], &Ws[buf][r0 * 32 + c0]);
    async_load_b128(&W[(size_t)(n0 + r0 + 64) * K + k0 + c0], &Ws[buf][(r0 + 64) * 32 + c0]);
  };

  v8f acc[8] = {};
  const int NT = K / 32;
  issue_tile(0, 0);

  for (int kt = 0; kt < NT; ++kt) {
    const int cur = kt & 1;
    wait_async_zero();          // my async writes to As/Ws[cur] complete
    __syncthreads();            // everyone's complete; prev reads of cur^1 done
    if (kt + 1 < NT) issue_tile(cur ^ 1, (kt + 1) * 32);

    const h16* Ab = As[cur];
    const h16* Wb = Ws[cur];

    // A fragment (16x32), documented CDNA5 lane layout
    const int arow = wave * 16 + lh;
    const int kA   = hi ? 8 : 0;
    v8h alo = *(const v8h*)&Ab[arow * 32 + kA];
    v8h ahi = *(const v8h*)&Ab[arow * 32 + kA + 16];
    v16h afrag;
#pragma unroll
    for (int i = 0; i < 8; ++i) { afrag[i] = alo[i]; afrag[8 + i] = ahi[i]; }

    // batch all 8 B fragments, then 8 back-to-back WMMAs
    v16h bfr[8];
#pragma unroll
    for (int jn = 0; jn < 8; ++jn)
      bfr[jn] = __builtin_shufflevector(*(const v8h*)&Wb[(jn * 16 + lh) * 32 + (hi ? 8 : 0)], *(const v8h*)&Wb[(jn * 16 + lh) * 32 + (hi ? 8 : 0) + 16], 0,1,2,3,4,5,6,7,8,9,10,11,12,13,14,15);
#pragma unroll
    for (int jn = 0; jn < 8; ++jn)
      acc[jn] = __builtin_amdgcn_wmma_f32_16x16x32_f16(
          false, afrag, false, bfr[jn], (short)0, acc[jn], false, false);
  }

  // epilogue: C layout — lane = n within tile, VGPR r -> row r (+8 for hi half)
#pragma unroll
  for (int jn = 0; jn < 8; ++jn) {
    const int n = n0 + jn * 16 + lh;
    const float bn = bias[n];
#pragma unroll
    for (int r = 0; r < 8; ++r) {
      const int m = m0 + wave * 16 + r + hi * 8;
      const float val = acc[jn][r] + bn;
      if constexpr (MODE == 1) {
        h16* out = (h16*)outp;
        const int b = m >> 7, t = m & (SEQ - 1);
        const int h = n >> 6, d = n & (HDIM - 1);
        out[(size_t)((b * NHEADS + h) * SEQ + t) * HDIM + d] = (h16)val;
      } else {
        float* out = (float*)outp;
        out[(size_t)m * N + n] = val;
      }
    }
  }
}

// ---------------------------------------------------------------- RoPE (in-place on [bh, t, d] f16)
__global__ void rope_inplace(h16* __restrict__ Q, h16* __restrict__ Kk) {
  int idx = blockIdx.x * blockDim.x + threadIdx.x;  // BATCH*NHEADS*SEQ*32 total
  int j  = idx & 31;
  int t  = (idx >> 5) & (SEQ - 1);
  int bh = idx >> 12;
  float inv = __powf(10000.0f, -(float)j * (1.0f / 32.0f));
  float ang = (float)t * inv;
  float s, c;
  __sincosf(ang, &s, &c);
  size_t base = ((size_t)bh * SEQ + t) * HDIM;
  {
    float x1 = (float)Q[base + j], x2 = (float)Q[base + j + 32];
    Q[base + j]      = (h16)(x1 * c - x2 * s);
    Q[base + j + 32] = (h16)(x2 * c + x1 * s);
  }
  {
    float x1 = (float)Kk[base + j], x2 = (float)Kk[base + j + 32];
    Kk[base + j]      = (h16)(x1 * c - x2 * s);
    Kk[base + j + 32] = (h16)(x2 * c + x1 * s);
  }
}

// ---------------------------------------------------------------- attention
// One workgroup per (b,h). 8 waves, each owns 16 query rows.
__global__ __launch_bounds__(256)
void attention_wmma(const h16* __restrict__ Q, const h16* __restrict__ Kk,
                    const h16* __restrict__ V, h16* __restrict__ O) {
  __shared__ __align__(32) h16 Qs[SEQ * HDIM];   // 16KB
  __shared__ __align__(32) h16 Ks[SEQ * HDIM];   // 16KB
  __shared__ __align__(32) h16 Vt[HDIM * SEQ];   // 16KB, transposed [d][n]

  const int tid = threadIdx.x;
  const int wave = tid >> 5, lane = tid & 31;
  const int lh = lane & 15, hi = (lane >> 4) & 1;
  const int bh = blockIdx.x;
  const size_t base = (size_t)bh * SEQ * HDIM;

  // async Q/K tile loads (LDS-direct, ASYNCcnt); V transposed via scalar DS
#pragma unroll
  for (int rep = 0; rep < 4; ++rep) {
    int cid = tid + rep * 256;                      // 1024 chunks of 8 halves
    async_load_b128(&Q[base + (size_t)cid * 8],  &Qs[cid * 8]);
    async_load_b128(&Kk[base + (size_t)cid * 8], &Ks[cid * 8]);
  }
  for (int idx = tid; idx < SEQ * HDIM; idx += 256) {
    int n = idx >> 6, d = idx & (HDIM - 1);
    Vt[d * SEQ + n] = V[base + idx];
  }
  wait_async_zero();
  __syncthreads();

  // S = Q K^T * scale; rows [wave*16, +16) x 128 cols
  v8f s[8] = {};
  const int arowb = wave * 16 + lh;
#pragma unroll
  for (int kc = 0; kc < 2; ++kc) {
    const int kA = kc * 32 + (hi ? 8 : 0);
    v8h alo = *(const v8h*)&Qs[arowb * HDIM + kA];
    v8h ahi = *(const v8h*)&Qs[arowb * HDIM + kA + 16];
    v16h afrag;
#pragma unroll
    for (int i = 0; i < 8; ++i) { afrag[i] = alo[i]; afrag[8 + i] = ahi[i]; }
    v16h bfr[8];
#pragma unroll
    for (int jn = 0; jn < 8; ++jn)
      bfr[jn] = __builtin_shufflevector(*(const v8h*)&Ks[(jn * 16 + lh) * HDIM + kc * 32 + (hi ? 8 : 0)], *(const v8h*)&Ks[(jn * 16 + lh) * HDIM + kc * 32 + (hi ? 8 : 0) + 16], 0,1,2,3,4,5,6,7,8,9,10,11,12,13,14,15);
#pragma unroll
    for (int jn = 0; jn < 8; ++jn)
      s[jn] = __builtin_amdgcn_wmma_f32_16x16x32_f16(
          false, afrag, false, bfr[jn], (short)0, s[jn], false, false);
  }

  // scale + causal mask + softmax in fp32 registers.
  // C layout: lane -> column n, VGPR r -> row (r + 8*hi). Each row's 16
  // columns of a tile live in one 16-lane half => reduce with xor 1,2,4,8.
  float rmax[8], rsum[8];
#pragma unroll
  for (int r = 0; r < 8; ++r) rmax[r] = -3.0e38f;
#pragma unroll
  for (int jn = 0; jn < 8; ++jn) {
    const int n = jn * 16 + lh;
#pragma unroll
    for (int r = 0; r < 8; ++r) {
      const int m = wave * 16 + r + hi * 8;
      float v = s[jn][r] * 0.125f;                  // 1/sqrt(64)
      if (n > m) v = -1.0e30f;
      s[jn][r] = v;
      rmax[r] = fmaxf(rmax[r], v);
    }
  }
#pragma unroll
  for (int r = 0; r < 8; ++r) {
#pragma unroll
    for (int off = 1; off <= 8; off <<= 1)
      rmax[r] = fmaxf(rmax[r], __shfl_xor(rmax[r], off, 32));
    rsum[r] = 0.0f;
  }
#pragma unroll
  for (int jn = 0; jn < 8; ++jn)
#pragma unroll
    for (int r = 0; r < 8; ++r) {
      float e = __expf(s[jn][r] - rmax[r]);
      s[jn][r] = e;
      rsum[r] += e;
    }
#pragma unroll
  for (int r = 0; r < 8; ++r) {
#pragma unroll
    for (int off = 1; off <= 8; off <<= 1)
      rsum[r] += __shfl_xor(rsum[r], off, 32);
    rsum[r] = 1.0f / rsum[r];
  }

  // Q/K LDS no longer needed -> reuse as per-wave P staging (16x128 f16 each)
  __syncthreads();
  h16* P = (wave < 4) ? &Qs[wave * 16 * SEQ] : &Ks[(wave - 4) * 16 * SEQ];
#pragma unroll
  for (int jn = 0; jn < 8; ++jn)
#pragma unroll
    for (int r = 0; r < 8; ++r) {
      const int row = r + hi * 8;
      P[row * SEQ + jn * 16 + lh] = (h16)(s[jn][r] * rsum[r]);
    }

  // O = P x V : M=16, K=128 (4 chunks), N=64 (4 tiles). Same-wave LDS RAW is
  // in-order per the DS pipeline; only this wave touches its P strip.
  v8f o[4] = {};
#pragma unroll
  for (int kc = 0; kc < 4; ++kc) {
    const int kA = kc * 32 + (hi ? 8 : 0);
    v8h alo = *(const v8h*)&P[lh * SEQ + kA];
    v8h ahi = *(const v8h*)&P[lh * SEQ + kA + 16];
    v16h afrag;
#pragma unroll
    for (int i = 0; i < 8; ++i) { afrag[i] = alo[i]; afrag[8 + i] = ahi[i]; }
    v16h bfr[4];
#pragma unroll
    for (int jd = 0; jd < 4; ++jd)
      bfr[jd] = __builtin_shufflevector(*(const v8h*)&Vt[(jd * 16 + lh) * SEQ + kc * 32 + (hi ? 8 : 0)], *(const v8h*)&Vt[(jd * 16 + lh) * SEQ + kc * 32 + (hi ? 8 : 0) + 16], 0,1,2,3,4,5,6,7,8,9,10,11,12,13,14,15);
#pragma unroll
    for (int jd = 0; jd < 4; ++jd)
      o[jd] = __builtin_amdgcn_wmma_f32_16x16x32_f16(
          false, afrag, false, bfr[jd], (short)0, o[jd], false, false);
  }

  // write O as f16 to [b, t, h*64+d] (row-major [MTOK, D_MODEL])
  const int b = bh >> 4, h = bh & (NHEADS - 1);
#pragma unroll
  for (int jd = 0; jd < 4; ++jd)
#pragma unroll
    for (int r = 0; r < 8; ++r) {
      const int t = wave * 16 + r + hi * 8;
      const int d = jd * 16 + lh;
      O[((size_t)(b * SEQ + t) * D_MODEL) + h * HDIM + d] = (h16)o[jd][r];
    }
}

// ---------------------------------------------------------------- launch
extern "C" void kernel_launch(void* const* d_in, const int* in_sizes, int n_in,
                              void* d_out, int out_size, void* d_ws, size_t ws_size,
                              hipStream_t stream) {
  (void)in_sizes; (void)n_in; (void)out_size; (void)ws_size;
  const float* x  = (const float*)d_in[0];
  const float* Wq = (const float*)d_in[1];
  const float* bq = (const float*)d_in[2];
  const float* Wk = (const float*)d_in[3];
  const float* bk = (const float*)d_in[4];
  const float* Wv = (const float*)d_in[5];
  const float* bv = (const float*)d_in[6];
  const float* Wo = (const float*)d_in[7];
  const float* bo = (const float*)d_in[8];
  float* out = (float*)d_out;

  char* ws = (char*)d_ws;
  const size_t SZ_X = (size_t)MTOK * D_MODEL;     // elements
  const size_t SZ_W = (size_t)D_MODEL * D_MODEL;
  size_t off = 0;
  h16* xh  = (h16*)(ws + off); off += SZ_X * 2;
  h16* wqh = (h16*)(ws + off); off += SZ_W * 2;
  h16* wkh = (h16*)(ws + off); off += SZ_W * 2;
  h16* wvh = (h16*)(ws + off); off += SZ_W * 2;
  h16* woh = (h16*)(ws + off); off += SZ_W * 2;
  h16* Qh  = (h16*)(ws + off); off += SZ_X * 2;   // [bh][t][d]
  h16* Kh  = (h16*)(ws + off); off += SZ_X * 2;
  h16* Vh  = (h16*)(ws + off); off += SZ_X * 2;
  h16* Oh  = (h16*)(ws + off); off += SZ_X * 2;   // [m][n] f16

  int n4 = (int)(SZ_X / 4);
  int w4 = (int)(SZ_W / 4);
  cvt_f32_to_f16<<<(n4 + 255) / 256, 256, 0, stream>>>(x,  xh,  n4);
  cvt_f32_to_f16<<<(w4 + 255) / 256, 256, 0, stream>>>(Wq, wqh, w4);
  cvt_f32_to_f16<<<(w4 + 255) / 256, 256, 0, stream>>>(Wk, wkh, w4);
  cvt_f32_to_f16<<<(w4 + 255) / 256, 256, 0, stream>>>(Wv, wvh, w4);
  cvt_f32_to_f16<<<(w4 + 255) / 256, 256, 0, stream>>>(Wo, woh, w4);

  dim3 ggrid(D_MODEL / 128, MTOK / 128);
  gemm_wmma_f16<1><<<ggrid, 256, 0, stream>>>(xh, wqh, bq, Qh, MTOK, D_MODEL, D_MODEL);
  gemm_wmma_f16<1><<<ggrid, 256, 0, stream>>>(xh, wkh, bk, Kh, MTOK, D_MODEL, D_MODEL);
  gemm_wmma_f16<1><<<ggrid, 256, 0, stream>>>(xh, wvh, bv, Vh, MTOK, D_MODEL, D_MODEL);

  rope_inplace<<<(BATCH * NHEADS * SEQ * 32) / 256, 256, 0, stream>>>(Qh, Kh);

  attention_wmma<<<BATCH * NHEADS, 256, 0, stream>>>(Qh, Kh, Vh, Oh);

  gemm_wmma_f16<2><<<ggrid, 256, 0, stream>>>(Oh, woh, bo, out, MTOK, D_MODEL, D_MODEL);
}
